// CompositeOneGRU_83958020702636
// MI455X (gfx1250) — compile-verified, matrix operand order checked by reading, not run on hardware
//
#include <hip/hip_runtime.h>
#include <hip/hip_bf16.h>
#include <math.h>

typedef __attribute__((ext_vector_type(16))) _Float16 v16h;
typedef __attribute__((ext_vector_type(8)))  float    v8f;

#define B_  512
#define N_  32
#define D_  300
#define R_  5
#define E_  256
#define KP  320      // K padded to 10 * 32
#define VG  50000
#define VS  20000

// ---------------------------------------------------------------------------
// Kernel 1: per-batch graph step, but only node 0's "proposed" row is needed.
// proposed0[b,:] = sum_r combined_r @ conv_W[r] + xs0 @ W0
// combined_r = sum_{e: dst==0, et==r} coeff_e * xs[src_e] + dinv0[r]^2 * xs0
// ---------------------------------------------------------------------------
__global__ __launch_bounds__(256)
void graph_step_kernel(const float* __restrict__ x,          // [B,N,D]
                       const int*   __restrict__ edge_index, // [B,2,E]
                       const int*   __restrict__ edge_type,  // [B,E]
                       const float* __restrict__ conv_W,     // [R,D,D]
                       const float* __restrict__ W0,         // [D,D]
                       float*       __restrict__ proposed0)  // [B,D]
{
    __shared__ int   cnt[N_][R_];
    __shared__ float dinv[N_][R_];
    __shared__ float vcomb[R_][D_];
    __shared__ float xs0[D_];
    __shared__ int   lsrc[E_];
    __shared__ int   let_[E_];
    __shared__ float lcf[E_];
    __shared__ int   nlist;

    const int b = blockIdx.x;
    const int t = threadIdx.x;
    const float* xb = x + (size_t)b * N_ * D_;

    for (int i = t; i < N_ * R_; i += 256) ((int*)cnt)[i] = 0;
    for (int i = t; i < R_ * D_; i += 256) ((float*)vcomb)[i] = 0.0f;
    if (t == 0) nlist = 0;
    __syncthreads();

    int esrc = 0, edst = 0, eet = 0;
    if (t < E_) {
        esrc = edge_index[(size_t)b * 2 * E_ + t];
        edst = edge_index[(size_t)b * 2 * E_ + E_ + t];
        eet  = edge_type [(size_t)b * E_ + t];
        atomicAdd(&cnt[edst][eet], 1);
    }
    __syncthreads();

    for (int i = t; i < N_ * R_; i += 256)
        ((float*)dinv)[i] = rsqrtf(1.0f + (float)((int*)cnt)[i]);
    for (int i = t; i < D_; i += 256) xs0[i] = xb[i];
    __syncthreads();

    // collect edges targeting node 0
    if (t < E_ && edst == 0) {
        int slot = atomicAdd(&nlist, 1);
        lsrc[slot] = esrc;
        let_[slot] = eet;
        lcf[slot]  = dinv[esrc][eet] * dinv[0][eet];
    }
    __syncthreads();

    const int ne = nlist;
    for (int i = 0; i < ne; ++i) {
        const float* xr = xb + (size_t)lsrc[i] * D_;
        const float  c  = lcf[i];
        const int    r  = let_[i];
        for (int d = t; d < D_; d += 256) vcomb[r][d] += c * xr[d];
    }
    __syncthreads();

    // self-loop term for node 0
    for (int d = t; d < D_; d += 256) {
        const float xv = xs0[d];
        #pragma unroll
        for (int r = 0; r < R_; ++r)
            vcomb[r][d] += dinv[0][r] * dinv[0][r] * xv;
    }
    __syncthreads();

    // 6 mat-vecs: proposed0[b, dout]
    for (int dout = t; dout < D_; dout += 256) {
        float acc = 0.0f;
        #pragma unroll
        for (int r = 0; r < R_; ++r) {
            const float* wr = conv_W + ((size_t)r * D_) * D_ + dout;
            for (int k = 0; k < D_; ++k) acc += vcomb[r][k] * wr[(size_t)k * D_];
        }
        const float* w0 = W0 + dout;
        for (int k = 0; k < D_; ++k) acc += xs0[k] * w0[(size_t)k * D_];
        proposed0[(size_t)b * D_ + dout] = acc;
    }
}

// ---------------------------------------------------------------------------
// Kernel 2: scalar-gated recurrence over B (sequential), relu, pack to f16
// padded to KP. Single block; each thread owns one feature dim.
// ---------------------------------------------------------------------------
__global__ __launch_bounds__(320)
void recurrence_pack_kernel(const float* __restrict__ proposed0, // [B,D]
                            const float* __restrict__ update_gate,
                            const float* __restrict__ memory0,   // [N,D] (row 0 used)
                            _Float16*    __restrict__ Xpack)     // [B,KP]
{
    const int d = threadIdx.x;
    const float g = update_gate[0];
    float mem = (d < D_) ? memory0[d] : 0.0f;
    for (int b = 0; b < B_; ++b) {
        float x1 = 0.0f;
        if (d < D_) {
            mem = g * proposed0[(size_t)b * D_ + d] + (1.0f - g) * mem;
            x1  = fmaxf(mem, 0.0f);
        }
        Xpack[(size_t)b * KP + d] = (_Float16)x1;
    }
}

// ---------------------------------------------------------------------------
// Kernel 3: pre-swizzle X into WMMA A-fragment lane order so each lane's
// 16 halfs are one contiguous 32B load.
// A 16x32 f16 layout: M = lane&15 ; K halves h<8 -> h else h+8 ; lanes>=16 +8.
// ---------------------------------------------------------------------------
__global__ __launch_bounds__(256)
void build_afrag_kernel(const _Float16* __restrict__ Xpack,   // [B,KP]
                        _Float16*       __restrict__ Afrag)   // [32][10][32][16]
{
    const int idx = blockIdx.x * 256 + threadIdx.x;
    if (idx >= (B_ / 16) * 10 * 32 * 16) return;
    const int h    = idx & 15;
    const int lane = (idx >> 4) & 31;
    const int kc   = (idx >> 9) % 10;
    const int mt   = idx / 5120;
    const int row  = mt * 16 + (lane & 15);
    const int k    = kc * 32 + (h < 8 ? h : h + 8) + (lane < 16 ? 0 : 8);
    Afrag[idx] = Xpack[(size_t)row * KP + k];
}

// ---------------------------------------------------------------------------
// Kernel 4: logits GEMM  out[512,V] = X @ W^T + bias  via v_wmma_f32_16x16x32_f16
// One block = 16 vocab columns. 8 waves x 4 M-tiles = all 512 rows.
// W tile converted f32->f16 into LDS (each W element read once globally).
// ---------------------------------------------------------------------------
__global__ __launch_bounds__(256)
void wmma_logits_kernel(const _Float16* __restrict__ Afrag,
                        const float*    __restrict__ W,    // [V,D]
                        const float*    __restrict__ bias, // [V]
                        float*          __restrict__ out,  // [B_,V]
                        int V)
{
    __shared__ _Float16 wtile[16 * KP]; // [n][k], 10 KB

    const int t = threadIdx.x;
    const int colbase = blockIdx.x * 16;

    for (int i = t; i < 16 * KP; i += 256) {
        const int n = i / KP, k = i % KP;
        const float v = (k < D_) ? W[(size_t)(colbase + n) * D_ + k] : 0.0f;
        wtile[i] = (_Float16)v;
    }
    __syncthreads();

    const int wave = t >> 5, lane = t & 31;
    const int n = lane & 15;
    const int kshift = (lane < 16) ? 0 : 8;

    v8f zero = {};
    v8f acc[4] = {zero, zero, zero, zero};

    union { v16h h; unsigned u[8]; } bf;
    const unsigned* wp = (const unsigned*)wtile;

    for (int kc = 0; kc < 10; ++kc) {
        const int bofs = n * KP + kc * 32 + kshift;   // even half-index
        #pragma unroll
        for (int j = 0; j < 8; ++j) {
            const int koff = (j < 4) ? 2 * j : 8 + 2 * j; // pairs {0..7,16..23}
            bf.u[j] = wp[(bofs + koff) >> 1];
        }
        #pragma unroll
        for (int m = 0; m < 4; ++m) {
            const int mt = wave * 4 + m;
            const v16h* ap =
                (const v16h*)(Afrag + ((size_t)(mt * 10 + kc) * 32 + lane) * 16);
            acc[m] = __builtin_amdgcn_wmma_f32_16x16x32_f16(
                false, *ap, false, bf.h, (short)0, acc[m], false, false);
        }
    }

    const float bv = bias[colbase + n];
    const int rsel = (lane < 16) ? 0 : 8;   // C layout: VGPR v -> M = v (+8)
    #pragma unroll
    for (int m = 0; m < 4; ++m) {
        const int rowb = wave * 64 + m * 16 + rsel;
        #pragma unroll
        for (int v = 0; v < 8; ++v)
            out[(size_t)(rowb + v) * V + colbase + n] = acc[m][v] + bv;
    }
}

// ---------------------------------------------------------------------------
// Kernel 5: fused per-row online max / log-sum-exp, in-place log_softmax.
// ---------------------------------------------------------------------------
__global__ __launch_bounds__(256)
void logsoftmax_rows_kernel(float* __restrict__ logits, int V)
{
    __shared__ float sm[256], ss[256];
    const int row = blockIdx.x, t = threadIdx.x;
    float* p = logits + (size_t)row * V;

    float m = -INFINITY, s = 0.0f;
    for (int v = t; v < V; v += 256) {
        const float xv = p[v];
        const float nm = fmaxf(m, xv);
        s = s * __expf(m - nm) + __expf(xv - nm);
        m = nm;
    }
    sm[t] = m; ss[t] = s;
    __syncthreads();
    for (int o = 128; o > 0; o >>= 1) {
        if (t < o) {
            const float m2 = sm[t + o], s2 = ss[t + o];
            const float nm = fmaxf(sm[t], m2);
            ss[t] = ss[t] * __expf(sm[t] - nm) + s2 * __expf(m2 - nm);
            sm[t] = nm;
        }
        __syncthreads();
    }
    const float c = sm[0] + __logf(ss[0]);
    for (int v = t; v < V; v += 256) p[v] -= c;
}

// ---------------------------------------------------------------------------
extern "C" void kernel_launch(void* const* d_in, const int* in_sizes, int n_in,
                              void* d_out, int out_size, void* d_ws, size_t ws_size,
                              hipStream_t stream)
{
    const float* x       = (const float*)d_in[0];
    const int*   eidx    = (const int*)  d_in[1];
    const int*   etype   = (const int*)  d_in[2];
    const float* conv_W  = (const float*)d_in[3];
    const float* W0      = (const float*)d_in[4];
    const float* gate    = (const float*)d_in[5];
    const float* glob_W  = (const float*)d_in[6];
    const float* glob_b  = (const float*)d_in[7];
    const float* sense_W = (const float*)d_in[8];
    const float* sense_b = (const float*)d_in[9];
    const float* memory0 = (const float*)d_in[10];
    float* out = (float*)d_out;

    char* ws = (char*)d_ws;
    float*    proposed0 = (float*)ws;                                  // 512*300 f32
    _Float16* Xpack     = (_Float16*)(ws + (size_t)B_ * D_ * sizeof(float)); // 512*320 f16
    _Float16* Afrag     = Xpack + (size_t)B_ * KP;                     // 163840 f16

    graph_step_kernel<<<B_, 256, 0, stream>>>(x, eidx, etype, conv_W, W0, proposed0);
    recurrence_pack_kernel<<<1, 320, 0, stream>>>(proposed0, gate, memory0, Xpack);
    build_afrag_kernel<<<640, 256, 0, stream>>>(Xpack, Afrag);

    wmma_logits_kernel<<<VG / 16, 256, 0, stream>>>(Afrag, glob_W, glob_b, out, VG);
    wmma_logits_kernel<<<VS / 16, 256, 0, stream>>>(Afrag, sense_W, sense_b,
                                                    out + (size_t)B_ * VG, VS);

    logsoftmax_rows_kernel<<<B_, 256, 0, stream>>>(out, VG);
    logsoftmax_rows_kernel<<<B_, 256, 0, stream>>>(out + (size_t)B_ * VG, VS);
}
